// PtHeadSelection_22531398435410
// MI455X (gfx1250) — matrix operand order repaired and checked by reading, not compiled, not run on hardware
//
#include <hip/hip_runtime.h>
#include <hip/hip_bf16.h>

typedef __attribute__((ext_vector_type(16))) __bf16 v16bf;
typedef __attribute__((ext_vector_type(8)))  __bf16 v8bf;
typedef __attribute__((ext_vector_type(8)))  float  v8f;

#define B_   8
#define S_   1024
#define DZ_  128
#define C_   8
#define R_   64
#define CR_  512
#define INV_REG 76.92307692307693f  /* 1/0.013 */

// Build a 16-element bf16 fragment from two contiguous 16-byte chunks.
static __device__ __forceinline__ v16bf ld_frag(const __bf16* p0, const __bf16* p1) {
  v8bf lo = *reinterpret_cast<const v8bf*>(p0);
  v8bf hi = *reinterpret_cast<const v8bf*>(p1);
  v16bf r;
#pragma unroll
  for (int i = 0; i < 8; ++i) { r[i] = lo[i]; r[8 + i] = hi[i]; }
  return r;
}

static __device__ __forceinline__ v8f wmma_bf16(v16bf a, v16bf b, v8f c) {
  return __builtin_amdgcn_wmma_f32_16x16x32_bf16(false, a, false, b, (short)0, c,
                                                 false, false);
}

// Wave-local LDS RAW fence: the P tile is wave-private, and a wave's DS ops
// are processed in order (CDNA5 §7.1), so a split-counter s_wait_dscnt is all
// that's needed between the D-layout stores and the A-layout reads — no
// block-wide barrier, keeping the 4 waves' WMMA pipelines decoupled.
static __device__ __forceinline__ void lds_wave_fence() {
  __builtin_amdgcn_wave_barrier();
  asm volatile("s_wait_dscnt 0x0" ::: "memory");
  __builtin_amdgcn_wave_barrier();
}

// ---------------------------------------------------------------------------
// Kernel 0: U,V (512x128 f32) -> transposed bf16 copies UT,VT (128x512)
// ---------------------------------------------------------------------------
__global__ __launch_bounds__(256) void k0_prep_uv(const float* __restrict__ U,
                                                  const float* __restrict__ V,
                                                  __bf16* __restrict__ UT,
                                                  __bf16* __restrict__ VT) {
  int idx = blockIdx.x * 256 + threadIdx.x;  // 0..65535
  int cr = idx / DZ_, d = idx % DZ_;
  UT[(size_t)d * CR_ + cr] = (__bf16)U[(size_t)cr * DZ_ + d];
  VT[(size_t)d * CR_ + cr] = (__bf16)V[(size_t)cr * DZ_ + d];
}

// ---------------------------------------------------------------------------
// Kernel 1: projections qz@U^T, qz@V^T + RoPE.  One block per (b,s).
// Outputs bf16: qu[bc][s][r], qv[bc][s][r], qvT[bc][r][s], quoT[bc][r][s]
// ---------------------------------------------------------------------------
__global__ __launch_bounds__(256) void k1_proj_rope(
    const float* __restrict__ qz, const float* __restrict__ cosp,
    const float* __restrict__ sinp, const float* __restrict__ U,
    const float* __restrict__ V, __bf16* __restrict__ qu,
    __bf16* __restrict__ qv, __bf16* __restrict__ qvT,
    __bf16* __restrict__ quoT) {
  __shared__ alignas(16) float z[DZ_];
  __shared__ alignas(16) float us[CR_];
  __shared__ alignas(16) float vs[CR_];
  int b = blockIdx.x / S_, s = blockIdx.x % S_;
  int tid = threadIdx.x;
  if (tid < DZ_) z[tid] = qz[((size_t)b * S_ + s) * DZ_ + tid];
  __syncthreads();
#pragma unroll
  for (int rep = 0; rep < 2; ++rep) {
    int cr = tid + rep * 256;
    const float* ur = U + (size_t)cr * DZ_;
    const float* vr = V + (size_t)cr * DZ_;
    float au = 0.f, av = 0.f;
#pragma unroll
    for (int k = 0; k < DZ_; k += 4) {
      float4 zk = *reinterpret_cast<const float4*>(&z[k]);
      float4 uk = *reinterpret_cast<const float4*>(ur + k);
      float4 vk = *reinterpret_cast<const float4*>(vr + k);
      au += zk.x * uk.x + zk.y * uk.y + zk.z * uk.z + zk.w * uk.w;
      av += zk.x * vk.x + zk.y * vk.y + zk.z * vk.z + zk.w * vk.w;
    }
    us[cr] = au;
    vs[cr] = av;
  }
  __syncthreads();
#pragma unroll
  for (int rep = 0; rep < 2; ++rep) {
    int cr = tid + rep * 256;
    int c = cr >> 6, r = cr & 63;
    float cv = cosp[((size_t)b * S_ + s) * R_ + r];
    float sv = sinp[((size_t)b * S_ + s) * R_ + r];
    int pr = c * R_ + ((r + 32) & 63);
    float sgn = (r < 32) ? -1.f : 1.f;  // rotate_half sign
    float u = us[cr], v = vs[cr];
    float ru = sgn * us[pr], rv = sgn * vs[pr];
    float qur = u * cv + ru * sv;   // apply()
    float quor = u * cv - ru * sv;  // apply_o()
    float qvr = v * cv + rv * sv;   // apply()
    int bc = b * C_ + c;
    size_t o_sr = ((size_t)bc * S_ + s) * R_ + r;
    size_t o_rs = ((size_t)bc * R_ + r) * S_ + s;
    qu[o_sr] = (__bf16)qur;
    qv[o_sr] = (__bf16)qvr;
    qvT[o_rs] = (__bf16)qvr;
    quoT[o_rs] = (__bf16)quor;
  }
}

// ---------------------------------------------------------------------------
// Kernel 2: row max per (bc, s).  Max-only pass (no exp, no rescale chain);
// the 1/REG_H scale is applied once at the end.  4 waves x 16 rows per block.
// ---------------------------------------------------------------------------
__global__ __launch_bounds__(128) void k2_rowmax(const __bf16* __restrict__ qu,
                                                 const __bf16* __restrict__ qv,
                                                 float* __restrict__ mbuf) {
  int bc = blockIdx.x;
  int sblk = blockIdx.y * 64;
  int wv = threadIdx.x >> 5, lane = threadIdx.x & 31;
  int lh = lane >> 4, col = lane & 15;
  const __bf16* quB = qu + (size_t)bc * S_ * R_;
  const __bf16* qvB = qv + (size_t)bc * S_ * R_;
  int arow = sblk + wv * 16 + col;
  const __bf16* ap = quB + (size_t)arow * R_;
  v16bf a0 = ld_frag(ap + lh * 16, ap + lh * 16 + 8);
  v16bf a1 = ld_frag(ap + 32 + lh * 16, ap + 40 + lh * 16);
  float m[8];
#pragma unroll
  for (int j = 0; j < 8; ++j) m[j] = -3.0e38f;
  for (int t0 = 0; t0 < S_; t0 += 16) {
    const __bf16* bp = qvB + (size_t)(t0 + col) * R_ + lh * 16;
    v16bf b0 = ld_frag(bp, bp + 8);
    v16bf b1 = ld_frag(bp + 32, bp + 40);
    v8f acc = {};
    acc = wmma_bf16(a0, b0, acc);
    acc = wmma_bf16(a1, b1, acc);
#pragma unroll
    for (int j = 0; j < 8; ++j) m[j] = fmaxf(m[j], acc[j]);
  }
#pragma unroll
  for (int j = 0; j < 8; ++j) {
#pragma unroll
    for (int mask = 1; mask < 16; mask <<= 1)
      m[j] = fmaxf(m[j], __shfl_xor(m[j], mask, 32));
  }
  if (col == 0) {
#pragma unroll
    for (int j = 0; j < 8; ++j) {
      int row = sblk + wv * 16 + j + lh * 8;
      mbuf[(size_t)bc * S_ + row] = m[j] * INV_REG;  // max of scaled scores
    }
  }
}

// ---------------------------------------------------------------------------
// Kernel 3: O1 = softmax(F) @ qv, accumulating the row sum l as a byproduct
// of the exp it already computes; writes 1/l for kernel 4.  apply_o RoPE ->
// qh1[b][s][c*R+r] bf16.
// ---------------------------------------------------------------------------
__global__ __launch_bounds__(128) void k3_av(
    const __bf16* __restrict__ qu, const __bf16* __restrict__ qv,
    const __bf16* __restrict__ qvT, const float* __restrict__ mbuf,
    float* __restrict__ linvbuf, const float* __restrict__ cosp,
    const float* __restrict__ sinp, __bf16* __restrict__ qh1) {
  __shared__ alignas(16) __bf16 pb[4][16][40];  // per-wave P tile, padded
  int bc = blockIdx.x;
  int b = bc >> 3, c = bc & 7;
  int sblk = blockIdx.y * 64;
  int wv = threadIdx.x >> 5, lane = threadIdx.x & 31;
  int lh = lane >> 4, col = lane & 15;
  const __bf16* quB = qu + (size_t)bc * S_ * R_;
  const __bf16* qvB = qv + (size_t)bc * S_ * R_;
  const __bf16* qvTB = qvT + (size_t)bc * R_ * S_;
  int arow = sblk + wv * 16 + col;
  const __bf16* ap = quB + (size_t)arow * R_;
  v16bf a0 = ld_frag(ap + lh * 16, ap + lh * 16 + 8);
  v16bf a1 = ld_frag(ap + 32 + lh * 16, ap + 40 + lh * 16);
  float mrow[8], lsum[8];
#pragma unroll
  for (int j = 0; j < 8; ++j) {
    int row = sblk + wv * 16 + j + lh * 8;
    mrow[j] = mbuf[(size_t)bc * S_ + row];
    lsum[j] = 0.f;
  }
  v8f o[4] = {{}, {}, {}, {}};
  for (int t0 = 0; t0 < S_; t0 += 32) {
#pragma unroll
    for (int tt = 0; tt < 2; ++tt) {
      const __bf16* bp = qvB + (size_t)(t0 + tt * 16 + col) * R_ + lh * 16;
      v16bf b0 = ld_frag(bp, bp + 8);
      v16bf b1 = ld_frag(bp + 32, bp + 40);
      v8f acc = {};
      acc = wmma_bf16(a0, b0, acc);
      acc = wmma_bf16(a1, b1, acc);
#pragma unroll
      for (int j = 0; j < 8; ++j) {
        float p = __expf(acc[j] * INV_REG - mrow[j]);
        lsum[j] += p;
        pb[wv][j + lh * 8][tt * 16 + col] = (__bf16)p;
      }
    }
    lds_wave_fence();  // wave-private tile: s_wait_dscnt only, no s_barrier
    const __bf16* pr = &pb[wv][col][0];
    v16bf pa = ld_frag(pr + lh * 16, pr + lh * 16 + 8);
#pragma unroll
    for (int nt = 0; nt < 4; ++nt) {
      const __bf16* bp = qvTB + (size_t)(nt * 16 + col) * S_ + t0 + lh * 16;
      v16bf bv = ld_frag(bp, bp + 8);
      o[nt] = wmma_bf16(pa, bv, o[nt]);
    }
    // next iteration's DS stores stay ordered behind this iteration's DS
    // reads (per-wave LDS ops are in-order); no wait needed here.
    __builtin_amdgcn_wave_barrier();
  }
  // finish the row sums across the 16 lanes sharing each row
  float linv[8];
#pragma unroll
  for (int j = 0; j < 8; ++j) {
#pragma unroll
    for (int mask = 1; mask < 16; mask <<= 1)
      lsum[j] += __shfl_xor(lsum[j], mask, 32);
    linv[j] = 1.0f / lsum[j];
  }
  if (col == 0) {
#pragma unroll
    for (int j = 0; j < 8; ++j) {
      int row = sblk + wv * 16 + j + lh * 8;
      linvbuf[(size_t)bc * S_ + row] = linv[j];
    }
  }
#pragma unroll
  for (int j = 0; j < 8; ++j) {
    int srow = sblk + wv * 16 + j + lh * 8;
    const float* crp = cosp + ((size_t)b * S_ + srow) * R_;
    const float* srp = sinp + ((size_t)b * S_ + srow) * R_;
    __bf16* outp = qh1 + ((size_t)b * S_ + srow) * CR_ + c * R_;
#pragma unroll
    for (int ntl = 0; ntl < 2; ++ntl) {
      int rlo = ntl * 16 + col, rhi = rlo + 32;
      float xlo = o[ntl][j] * linv[j];
      float xhi = o[ntl + 2][j] * linv[j];
      float olo = xlo * crp[rlo] + xhi * srp[rlo];  // apply_o, r<32
      float ohi = xhi * crp[rhi] - xlo * srp[rhi];  // apply_o, r>=32
      outp[rlo] = (__bf16)olo;
      outp[rhi] = (__bf16)ohi;
    }
  }
}

// ---------------------------------------------------------------------------
// Kernel 4: O2 = softmax(F)^T @ quo, then apply RoPE -> qh2[b][s][c*R+r] bf16
// Scores computed transposed directly (A=qv rows, B=qu rows); per-t softmax
// normalization (m_t, 1/l_t from k3) is a per-lane scale in the D layout.
// ---------------------------------------------------------------------------
__global__ __launch_bounds__(128) void k4_atu(
    const __bf16* __restrict__ qu, const __bf16* __restrict__ qv,
    const __bf16* __restrict__ quoT, const float* __restrict__ mbuf,
    const float* __restrict__ linvbuf, const float* __restrict__ cosp,
    const float* __restrict__ sinp, __bf16* __restrict__ qh2) {
  __shared__ alignas(16) __bf16 pb[4][16][40];
  int bc = blockIdx.x;
  int b = bc >> 3, c = bc & 7;
  int sblk = blockIdx.y * 64;
  int wv = threadIdx.x >> 5, lane = threadIdx.x & 31;
  int lh = lane >> 4, col = lane & 15;
  const __bf16* quB = qu + (size_t)bc * S_ * R_;
  const __bf16* qvB = qv + (size_t)bc * S_ * R_;
  const __bf16* quoTB = quoT + (size_t)bc * R_ * S_;
  int arow = sblk + wv * 16 + col;
  const __bf16* ap = qvB + (size_t)arow * R_;  // A = qv rows (scores^T)
  v16bf a0 = ld_frag(ap + lh * 16, ap + lh * 16 + 8);
  v16bf a1 = ld_frag(ap + 32 + lh * 16, ap + 40 + lh * 16);
  v8f o[4] = {{}, {}, {}, {}};
  for (int t0 = 0; t0 < S_; t0 += 32) {
#pragma unroll
    for (int tt = 0; tt < 2; ++tt) {
      int t = t0 + tt * 16 + col;
      const __bf16* bp = quB + (size_t)t * R_ + lh * 16;  // B cols = t rows of qu
      v16bf b0 = ld_frag(bp, bp + 8);
      v16bf b1 = ld_frag(bp + 32, bp + 40);
      v8f acc = {};
      acc = wmma_bf16(a0, b0, acc);
      acc = wmma_bf16(a1, b1, acc);
      float mt = mbuf[(size_t)bc * S_ + t];
      float lti = linvbuf[(size_t)bc * S_ + t];
#pragma unroll
      for (int j = 0; j < 8; ++j) {
        float p = __expf(acc[j] * INV_REG - mt) * lti;
        pb[wv][j + lh * 8][tt * 16 + col] = (__bf16)p;
      }
    }
    lds_wave_fence();  // wave-private tile: s_wait_dscnt only, no s_barrier
    const __bf16* pr = &pb[wv][col][0];
    v16bf pa = ld_frag(pr + lh * 16, pr + lh * 16 + 8);
#pragma unroll
    for (int nt = 0; nt < 4; ++nt) {
      const __bf16* bp = quoTB + (size_t)(nt * 16 + col) * S_ + t0 + lh * 16;
      v16bf bv = ld_frag(bp, bp + 8);
      o[nt] = wmma_bf16(pa, bv, o[nt]);
    }
    __builtin_amdgcn_wave_barrier();
  }
#pragma unroll
  for (int j = 0; j < 8; ++j) {
    int srow = sblk + wv * 16 + j + lh * 8;
    const float* crp = cosp + ((size_t)b * S_ + srow) * R_;
    const float* srp = sinp + ((size_t)b * S_ + srow) * R_;
    __bf16* outp = qh2 + ((size_t)b * S_ + srow) * CR_ + c * R_;
#pragma unroll
    for (int ntl = 0; ntl < 2; ++ntl) {
      int rlo = ntl * 16 + col, rhi = rlo + 32;
      float xlo = o[ntl][j];
      float xhi = o[ntl + 2][j];
      float olo = xlo * crp[rlo] - xhi * srp[rlo];  // apply, r<32
      float ohi = xhi * crp[rhi] + xlo * srp[rhi];  // apply, r>=32
      outp[rlo] = (__bf16)olo;
      outp[rhi] = (__bf16)ohi;
    }
  }
}

// ---------------------------------------------------------------------------
// Kernel 5: out = qh1 @ U + qh2 @ V  (8192 x 128 x 512 bf16 WMMA, f32 out)
// ---------------------------------------------------------------------------
__global__ __launch_bounds__(128) void k5_out(const __bf16* __restrict__ qh1,
                                              const __bf16* __restrict__ qh2,
                                              const __bf16* __restrict__ UT,
                                              const __bf16* __restrict__ VT,
                                              float* __restrict__ out) {
  int wv = threadIdx.x >> 5, lane = threadIdx.x & 31;
  int lh = lane >> 4, col = lane & 15;
  int rowbase = blockIdx.x * 64 + wv * 16;
  int arow = rowbase + col;
  v8f acc[8] = {{}, {}, {}, {}, {}, {}, {}, {}};
  const __bf16* srcs[2] = {qh1, qh2};
  const __bf16* wts[2] = {UT, VT};
#pragma unroll
  for (int mtx = 0; mtx < 2; ++mtx) {
    const __bf16* ab = srcs[mtx] + (size_t)arow * CR_;
    const __bf16* wb = wts[mtx];
    for (int k0 = 0; k0 < CR_; k0 += 32) {
      v16bf a = ld_frag(ab + k0 + lh * 16, ab + k0 + lh * 16 + 8);
#pragma unroll
      for (int nt = 0; nt < 8; ++nt) {
        const __bf16* bp = wb + (size_t)(nt * 16 + col) * CR_ + k0 + lh * 16;
        v16bf bv = ld_frag(bp, bp + 8);
        acc[nt] = wmma_bf16(a, bv, acc[nt]);
      }
    }
  }
#pragma unroll
  for (int nt = 0; nt < 8; ++nt) {
#pragma unroll
    for (int j = 0; j < 8; ++j) {
      int row = rowbase + j + lh * 8;
      out[(size_t)row * DZ_ + nt * 16 + col] = acc[nt][j];
    }
  }
}

// ---------------------------------------------------------------------------
extern "C" void kernel_launch(void* const* d_in, const int* in_sizes, int n_in,
                              void* d_out, int out_size, void* d_ws,
                              size_t ws_size, hipStream_t stream) {
  const float* qz = (const float*)d_in[0];
  const float* cosp = (const float*)d_in[1];
  const float* sinp = (const float*)d_in[2];
  const float* U = (const float*)d_in[3];
  const float* V = (const float*)d_in[4];
  char* ws = (char*)d_ws;
  // bf16 planes: B*C*S*R * 2B = 8 MiB each
  __bf16* qu = (__bf16*)(ws + 0);
  __bf16* qv = (__bf16*)(ws + 8388608);
  __bf16* qvT = (__bf16*)(ws + 16777216);
  __bf16* quoT = (__bf16*)(ws + 25165824);
  __bf16* qh1 = (__bf16*)(ws + 33554432);
  __bf16* qh2 = (__bf16*)(ws + 41943040);
  __bf16* UT = (__bf16*)(ws + 50331648);
  __bf16* VT = (__bf16*)(ws + 50462720);
  float* mbuf = (float*)(ws + 50593792);
  float* linvbuf = (float*)(ws + 50855936);
  float* out = (float*)d_out;

  k0_prep_uv<<<256, 256, 0, stream>>>(U, V, UT, VT);
  k1_proj_rope<<<B_ * S_, 256, 0, stream>>>(qz, cosp, sinp, U, V, qu, qv, qvT,
                                            quoT);
  dim3 g2(B_ * C_, S_ / 64);
  k2_rowmax<<<g2, 128, 0, stream>>>(qu, qv, mbuf);
  k3_av<<<g2, 128, 0, stream>>>(qu, qv, qvT, mbuf, linvbuf, cosp, sinp, qh1);
  k4_atu<<<g2, 128, 0, stream>>>(qu, qv, quoT, mbuf, linvbuf, cosp, sinp, qh2);
  k5_out<<<B_ * S_ / 64, 128, 0, stream>>>(qh1, qh2, UT, VT, out);
}
